// DITBlock_44822278701406
// MI455X (gfx1250) — matrix operand order repaired and checked by reading, not compiled, not run on hardware
//
#include <hip/hip_runtime.h>

// ---------------------------------------------------------------------------
// DiT block for MI455X (gfx1250, wave32, WMMA).  All GEMMs + both attention
// einsums run on v_wmma_f32_16x16x32_f16 (f16 in, f32 accumulate).
// Softmax row-reductions use DPP16 butterflies (VALU) instead of ds_bpermute.
// Shapes: B=4 N=2048 M=512 E=512 CD=256 H=8 DH=64 MH=1024.
// ---------------------------------------------------------------------------

typedef _Float16 half_t;
typedef __attribute__((ext_vector_type(16))) _Float16 v16h;
typedef __attribute__((ext_vector_type(8)))  float    v8f;

#define B_   4
#define N_   2048
#define M_   512
#define E_   512
#define CD_  256
#define H_   8
#define DH_  64
#define MH_  1024

// ------------------------------ WMMA helpers -------------------------------
// A fragment (16x32 f16): lane m=l&15, g=l>>4; element e: K = (e<8 ? g*8+e
//                                                              : 16+g*8+(e-8))
// B fragment (32x16 f16): lane n=l&15, g=l>>4; element e: same K as A
// C/D (16x16 f32): lane col n=l&15, VGPR r -> row = r + 8*(l>>4)

__device__ __forceinline__ v8f wmma32(v16h a, v16h b, v8f c) {
  return __builtin_amdgcn_wmma_f32_16x16x32_f16(
      /*neg_a=*/false, a, /*neg_b=*/false, b,
      /*c_mod=*/(short)0, c, /*reuse_a=*/false, /*reuse_b=*/false);
}

__device__ __forceinline__ v16h load_afrag(const half_t* base, int stride) {
  const int lane = threadIdx.x & 31;
  const int m = lane & 15, g = lane >> 4;
  const half_t* p0 = base + m * stride + g * 8;
  const half_t* p1 = base + m * stride + 16 + g * 8;
  union { v16h v; uint4 q[2]; } u;
  u.q[0] = *reinterpret_cast<const uint4*>(p0);
  u.q[1] = *reinterpret_cast<const uint4*>(p1);
  return u.v;
}

__device__ __forceinline__ v16h load_bfrag(const half_t* base, int stride) {
  const int lane = threadIdx.x & 31;
  const int n = lane & 15, g = lane >> 4;
  const half_t* p = base + n * stride + g * 8;
  union { v16h v; uint4 q[2]; } u;
  u.q[0] = *reinterpret_cast<const uint4*>(p);
  u.q[1] = *reinterpret_cast<const uint4*>(p + 16);
  return u.v;
}

// ---------------------- DPP16 16-lane butterflies --------------------------
// quad_perm(1,0,3,2)=0xB1 (xor1), quad_perm(2,3,0,1)=0x4E (xor2),
// ROW_HALF_MIRROR=0x141 (pairs quads within 8), ROW_MIRROR=0x140 (pairs 8s
// within the 16-lane DPP row).  All lanes end with the group reduction; the
// two 16-lane halves of the wave stay independent (matches D-layout rows).
template <int CTRL>
__device__ __forceinline__ float dpp_f(float x) {
  return __int_as_float(__builtin_amdgcn_update_dpp(
      0, __float_as_int(x), CTRL, 0xF, 0xF, true));
}
__device__ __forceinline__ float row16_max(float v) {
  v = fmaxf(v, dpp_f<0xB1>(v));
  v = fmaxf(v, dpp_f<0x4E>(v));
  v = fmaxf(v, dpp_f<0x141>(v));
  v = fmaxf(v, dpp_f<0x140>(v));
  return v;
}
__device__ __forceinline__ float row16_sum(float v) {
  v += dpp_f<0xB1>(v);
  v += dpp_f<0x4E>(v);
  v += dpp_f<0x141>(v);
  v += dpp_f<0x140>(v);
  return v;
}

// ------------------------------ small kernels ------------------------------
__global__ __launch_bounds__(256) void cvt_f32_to_f16(const float* __restrict__ s,
                                                      half_t* __restrict__ d, int n) {
  int i = blockIdx.x * 256 + threadIdx.x;
  if (i < n) d[i] = (half_t)s[i];
}

// LayerNorm over rows of width 512 (no affine), fp32 in, f16 out.
__global__ __launch_bounds__(256) void ln512_f16(const float* __restrict__ x,
                                                 half_t* __restrict__ y) {
  const int lane = threadIdx.x & 31;
  const int wave = threadIdx.x >> 5;
  const size_t row = (size_t)blockIdx.x * 8 + wave;
  const float* xr = x + row * 512;
  float v[16];
  float s = 0.f;
#pragma unroll
  for (int i = 0; i < 16; ++i) { v[i] = xr[lane + 32 * i]; s += v[i]; }
#pragma unroll
  for (int off = 1; off < 32; off <<= 1) s += __shfl_xor(s, off, 32);
  const float mean = s * (1.f / 512.f);
  float var = 0.f;
#pragma unroll
  for (int i = 0; i < 16; ++i) { float d = v[i] - mean; var += d * d; }
#pragma unroll
  for (int off = 1; off < 32; off <<= 1) var += __shfl_xor(var, off, 32);
  const float rstd = rsqrtf(var * (1.f / 512.f) + 1e-6f);
  half_t* yr = y + row * 512;
#pragma unroll
  for (int i = 0; i < 16; ++i) yr[lane + 32 * i] = (half_t)((v[i] - mean) * rstd);
}

// ------------------------------ WMMA GEMM ----------------------------------
// out[M,N] = epilogue( A[M,K] * W[N,K]^T + bias )  (+ fp32 residual)
// Block: 256 thr = 8 waves (2m x 4n), tile 128x128, BK=32.
// Wave tile 64x32: 8 WMMA per K-step, B frags reused across 4 A subtiles.
template <bool RELU, bool OUT_F16>
__global__ __launch_bounds__(256) void gemm_wmma_f16(
    const half_t* __restrict__ A, const half_t* __restrict__ W,
    const float* __restrict__ bias, const float* __restrict__ resid,
    void* __restrict__ outp, int M, int N, int K) {
  __shared__ half_t As[128 * 32];
  __shared__ half_t Bs[128 * 32];
  const int tid = threadIdx.x;
  const int wave = tid >> 5;
  const int wm = wave >> 2;  // 0..1 : 64-row slab
  const int wn = wave & 3;   // 0..3 : 32-col slab
  const int m0 = blockIdx.y * 128;
  const int n0 = blockIdx.x * 128;

  v8f acc[4][2] = {};
  for (int k0 = 0; k0 < K; k0 += 32) {
    {  // stage A tile: 128x32 halves, 16 halves (2x uint4) per thread
      const int row = tid >> 1, seg = tid & 1;
      const half_t* gp = A + (size_t)(m0 + row) * K + k0 + seg * 16;
      uint4* sp = reinterpret_cast<uint4*>(&As[row * 32 + seg * 16]);
      const uint4* g4 = reinterpret_cast<const uint4*>(gp);
      sp[0] = g4[0];
      sp[1] = g4[1];
    }
    {  // stage W tile: 128x32 halves, same pattern
      const int row = tid >> 1, seg = tid & 1;
      const half_t* gp = W + (size_t)(n0 + row) * K + k0 + seg * 16;
      uint4* sp = reinterpret_cast<uint4*>(&Bs[row * 32 + seg * 16]);
      const uint4* g4 = reinterpret_cast<const uint4*>(gp);
      sp[0] = g4[0];
      sp[1] = g4[1];
    }
    __syncthreads();
    v16h b0 = load_bfrag(Bs + (wn * 32) * 32, 32);
    v16h b1 = load_bfrag(Bs + (wn * 32 + 16) * 32, 32);
#pragma unroll
    for (int mi = 0; mi < 4; ++mi) {
      v16h a = load_afrag(As + (wm * 64 + mi * 16) * 32, 32);
      acc[mi][0] = wmma32(a, b0, acc[mi][0]);
      acc[mi][1] = wmma32(a, b1, acc[mi][1]);
    }
    __syncthreads();
  }

  const int lane = tid & 31;
  const int nl = lane & 15, g = lane >> 4;
#pragma unroll
  for (int mi = 0; mi < 4; ++mi) {
#pragma unroll
    for (int ni = 0; ni < 2; ++ni) {
      const int col = n0 + wn * 32 + ni * 16 + nl;
      const float bv = bias ? bias[col] : 0.f;
#pragma unroll
      for (int r = 0; r < 8; ++r) {
        const int row = m0 + wm * 64 + mi * 16 + r + 8 * g;
        float v = acc[mi][ni][r] + bv;
        if (RELU) v = fmaxf(v, 0.f);
        if (resid) v += resid[(size_t)row * N + col];
        if (OUT_F16)
          ((half_t*)outp)[(size_t)row * N + col] = (half_t)v;
        else
          ((float*)outp)[(size_t)row * N + col] = v;
      }
    }
  }
}

// --------------------------- Flash attention -------------------------------
// Per block: 128 query rows (8 waves x 16 rows), one (batch, head) in grid.y.
// K tile staged raw [32 keys][64 dh]; V tile staged transposed [64 dh][32 k]
// with packed b32 LDS stores.  S = Q K^T via WMMA, online softmax in the
// D-register layout with DPP16 reductions, P bounced through a per-wave LDS
// tile to become an A fragment for the P*V WMMAs.
__global__ __launch_bounds__(256) void flash_wmma(
    const half_t* __restrict__ Q, const half_t* __restrict__ K,
    const half_t* __restrict__ V, half_t* __restrict__ O,
    int Lq, int Lk, int sq, int sk, int sv, int so, float scale) {
  __shared__ half_t Ks[32 * 64];       // [key][dh]
  __shared__ half_t Vt[64 * 32];       // [dh][key]
  __shared__ half_t Ps[8 * 16 * 32];   // per-wave 16x32 P tile

  const int tid = threadIdx.x, lane = tid & 31, wave = tid >> 5;
  const int bh = blockIdx.y;
  const int b = bh >> 3, h = bh & 7;   // H_ == 8
  const int q0 = blockIdx.x * 128 + wave * 16;
  const int m = lane & 15, g = lane >> 4;

  const half_t* Qb = Q + (size_t)b * Lq * sq + h * DH_;
  const half_t* Kb = K + (size_t)b * Lk * sk + h * DH_;
  const half_t* Vb = V + (size_t)b * Lk * sv + h * DH_;
  half_t*       Ob = O + (size_t)b * Lq * so + h * DH_;

  // Loop-invariant Q fragments: dh chunks [0,32) and [32,64)
  v16h qf[2];
#pragma unroll
  for (int c = 0; c < 2; ++c) {
    const half_t* p0 = Qb + (size_t)(q0 + m) * sq + c * 32 + g * 8;
    const half_t* p1 = Qb + (size_t)(q0 + m) * sq + c * 32 + 16 + g * 8;
    union { v16h v; uint4 q[2]; } u;
    u.q[0] = *reinterpret_cast<const uint4*>(p0);
    u.q[1] = *reinterpret_cast<const uint4*>(p1);
    qf[c] = u.v;
  }

  v8f o[4] = {};
  float rm[8], rl[8];
#pragma unroll
  for (int r = 0; r < 8; ++r) { rm[r] = -1e30f; rl[r] = 0.f; }

  half_t* pw = Ps + wave * (16 * 32);

  for (int kt = 0; kt < Lk; kt += 32) {
    {  // stage K tile: 2048 halves, 8 per thread (uint4)
      const int idx = tid * 8, kr = idx >> 6, dc = idx & 63;
      const half_t* gp = Kb + (size_t)(kt + kr) * sk + dc;
      *reinterpret_cast<uint4*>(&Ks[kr * 64 + dc]) =
          *reinterpret_cast<const uint4*>(gp);
    }
    {  // stage V transposed: 2 k-rows x 4 dh per thread, packed b32 stores
      const int kr = (tid >> 4) * 2;
      const int dc = (tid & 15) * 4;
      const half_t* g0 = Vb + (size_t)(kt + kr) * sv + dc;
      const half_t* g1 = g0 + sv;
#pragma unroll
      for (int i = 0; i < 4; ++i) {
        union { half_t h[2]; unsigned u; } p;
        p.h[0] = g0[i];
        p.h[1] = g1[i];
        *reinterpret_cast<unsigned*>(&Vt[(dc + i) * 32 + kr]) = p.u;
      }
    }
    __syncthreads();

    // S tiles: key group 0 (keys kt+0..15) and 1 (kt+16..31)
    v8f s0 = {}, s1 = {};
    s0 = wmma32(qf[0], load_bfrag(Ks + 0, 64), s0);
    s0 = wmma32(qf[1], load_bfrag(Ks + 32, 64), s0);
    s1 = wmma32(qf[0], load_bfrag(Ks + 16 * 64, 64), s1);
    s1 = wmma32(qf[1], load_bfrag(Ks + 16 * 64 + 32, 64), s1);

    // Online softmax (per row r+8g; DPP16 reductions across the 16 lanes)
    float p0[8], p1[8];
#pragma unroll
    for (int r = 0; r < 8; ++r) {
      const float a = s0[r] * scale, bb = s1[r] * scale;
      const float mx = row16_max(fmaxf(a, bb));
      const float nm = fmaxf(rm[r], mx);
      const float corr = __expf(rm[r] - nm);
      rm[r] = nm;
      const float e0 = __expf(a - nm), e1 = __expf(bb - nm);
      rl[r] = rl[r] * corr + row16_sum(e0 + e1);
      p0[r] = e0; p1[r] = e1;
      o[0][r] *= corr; o[1][r] *= corr; o[2][r] *= corr; o[3][r] *= corr;
    }

    // D-layout -> A-layout via per-wave LDS bounce (wave-local, DS in-order)
#pragma unroll
    for (int r = 0; r < 8; ++r) {
      pw[(r + 8 * g) * 32 + m]      = (half_t)p0[r];
      pw[(r + 8 * g) * 32 + 16 + m] = (half_t)p1[r];
    }
    v16h pf = load_afrag(pw, 32);

    o[0] = wmma32(pf, load_bfrag(Vt + 0 * 16 * 32, 32), o[0]);
    o[1] = wmma32(pf, load_bfrag(Vt + 1 * 16 * 32, 32), o[1]);
    o[2] = wmma32(pf, load_bfrag(Vt + 2 * 16 * 32, 32), o[2]);
    o[3] = wmma32(pf, load_bfrag(Vt + 3 * 16 * 32, 32), o[3]);
    __syncthreads();
  }

#pragma unroll
  for (int r = 0; r < 8; ++r) {
    const float inv = 1.f / rl[r];
    half_t* op = Ob + (size_t)(q0 + r + 8 * g) * so;
    op[0 * 16 + m] = (half_t)(o[0][r] * inv);
    op[1 * 16 + m] = (half_t)(o[1][r] * inv);
    op[2 * 16 + m] = (half_t)(o[2][r] * inv);
    op[3 * 16 + m] = (half_t)(o[3][r] * inv);
  }
}

// ------------------------------- launcher ----------------------------------
extern "C" void kernel_launch(void* const* d_in, const int* in_sizes, int n_in,
                              void* d_out, int out_size, void* d_ws, size_t ws_size,
                              hipStream_t stream) {
  (void)in_sizes; (void)n_in; (void)out_size; (void)ws_size;
  const float* cond = (const float*)d_in[0];
  const float* x_in = (const float*)d_in[1];
  const float* Wqkv = (const float*)d_in[2];
  const float* bqkv = (const float*)d_in[3];
  const float* Wo   = (const float*)d_in[4];
  const float* bo   = (const float*)d_in[5];
  const float* Wcq  = (const float*)d_in[6];
  const float* Wck  = (const float*)d_in[7];
  const float* Wcv  = (const float*)d_in[8];
  const float* Wco  = (const float*)d_in[9];
  const float* bco  = (const float*)d_in[10];
  const float* W1   = (const float*)d_in[11];
  const float* b1   = (const float*)d_in[12];
  const float* W2   = (const float*)d_in[13];
  const float* b2   = (const float*)d_in[14];

  const int BN = B_ * N_;   // 8192 token rows
  const int BM = B_ * M_;   // 2048 cond rows

  size_t off = 0;
  auto alloc = [&](size_t bytes) -> void* {
    void* p = (char*)d_ws + off;
    off += (bytes + 255) & ~(size_t)255;
    return p;
  };
  half_t* wqkv_h = (half_t*)alloc((size_t)3 * E_ * E_ * 2);
  half_t* wo_h   = (half_t*)alloc((size_t)E_ * E_ * 2);
  half_t* wcq_h  = (half_t*)alloc((size_t)E_ * E_ * 2);
  half_t* wck_h  = (half_t*)alloc((size_t)E_ * CD_ * 2);
  half_t* wcv_h  = (half_t*)alloc((size_t)E_ * CD_ * 2);
  half_t* wco_h  = (half_t*)alloc((size_t)E_ * E_ * 2);
  half_t* w1_h   = (half_t*)alloc((size_t)MH_ * E_ * 2);
  half_t* w2_h   = (half_t*)alloc((size_t)E_ * MH_ * 2);
  half_t* xn_h   = (half_t*)alloc((size_t)BN * E_ * 2);        // reused for all 3 LNs
  half_t* big    = (half_t*)alloc((size_t)BN * 3 * E_ * 2);    // qkv / cq,ck,cv,cond / h1
  half_t* attn_h = (half_t*)alloc((size_t)BN * E_ * 2);        // attention outputs
  float*  x1     = (float*)alloc((size_t)BN * E_ * 4);
  float*  x2     = (float*)alloc((size_t)BN * E_ * 4);

  // region reuse inside `big` after self-attention consumes qkv
  half_t* qkv_h  = big;
  half_t* cq_h   = big;
  half_t* ck_h   = big + (size_t)BN * E_;
  half_t* cv_h   = ck_h + (size_t)BM * E_;
  half_t* cond_h = cv_h + (size_t)BM * E_;
  half_t* h1_h   = big;  // cq/ck/cv/cond dead before MLP

  auto cvt = [&](const float* s, half_t* d, int n) {
    cvt_f32_to_f16<<<(n + 255) / 256, 256, 0, stream>>>(s, d, n);
  };
  cvt(Wqkv, wqkv_h, 3 * E_ * E_);
  cvt(Wo, wo_h, E_ * E_);
  cvt(Wcq, wcq_h, E_ * E_);
  cvt(Wck, wck_h, E_ * CD_);
  cvt(Wcv, wcv_h, E_ * CD_);
  cvt(Wco, wco_h, E_ * E_);
  cvt(W1, w1_h, MH_ * E_);
  cvt(W2, w2_h, E_ * MH_);

  const float scale = 0.125f;  // 1/sqrt(64)

  // ---- self-attention ----
  ln512_f16<<<BN / 8, 256, 0, stream>>>(x_in, xn_h);
  gemm_wmma_f16<false, true><<<dim3(3 * E_ / 128, BN / 128), 256, 0, stream>>>(
      xn_h, wqkv_h, bqkv, nullptr, qkv_h, BN, 3 * E_, E_);
  flash_wmma<<<dim3(N_ / 128, B_ * H_), 256, 0, stream>>>(
      qkv_h, qkv_h + E_, qkv_h + 2 * E_, attn_h,
      N_, N_, 3 * E_, 3 * E_, 3 * E_, E_, scale);
  gemm_wmma_f16<false, false><<<dim3(E_ / 128, BN / 128), 256, 0, stream>>>(
      attn_h, wo_h, bo, x_in, x1, BN, E_, E_);

  // ---- cross-attention ----
  ln512_f16<<<BN / 8, 256, 0, stream>>>(x1, xn_h);
  cvt(cond, cond_h, BM * CD_);
  gemm_wmma_f16<false, true><<<dim3(E_ / 128, BN / 128), 256, 0, stream>>>(
      xn_h, wcq_h, nullptr, nullptr, cq_h, BN, E_, E_);
  gemm_wmma_f16<false, true><<<dim3(E_ / 128, BM / 128), 256, 0, stream>>>(
      cond_h, wck_h, nullptr, nullptr, ck_h, BM, E_, CD_);
  gemm_wmma_f16<false, true><<<dim3(E_ / 128, BM / 128), 256, 0, stream>>>(
      cond_h, wcv_h, nullptr, nullptr, cv_h, BM, E_, CD_);
  flash_wmma<<<dim3(N_ / 128, B_ * H_), 256, 0, stream>>>(
      cq_h, ck_h, cv_h, attn_h, N_, M_, E_, E_, E_, E_, scale);
  gemm_wmma_f16<false, false><<<dim3(E_ / 128, BN / 128), 256, 0, stream>>>(
      attn_h, wco_h, bco, x1, x2, BN, E_, E_);

  // ---- MLP (ReLU after both layers; residual after final ReLU) ----
  ln512_f16<<<BN / 8, 256, 0, stream>>>(x2, xn_h);
  gemm_wmma_f16<true, true><<<dim3(MH_ / 128, BN / 128), 256, 0, stream>>>(
      xn_h, w1_h, b1, nullptr, h1_h, BN, MH_, E_);
  gemm_wmma_f16<true, false><<<dim3(E_ / 128, BN / 128), 256, 0, stream>>>(
      h1_h, w2_h, b2, x2, (float*)d_out, BN, E_, MH_);
}
